// Transformer_67250597920935
// MI455X (gfx1250) — matrix operand order repaired and checked by reading, not compiled, not run on hardware
//
#include <hip/hip_runtime.h>

// ---------------------------------------------------------------------------
// Transformer forward for MI455X (gfx1250, wave32, WMMA).
// All GEMM-shaped work (QKV/out projections, FFN, QK^T, PV, vocab logits)
// runs through v_wmma_f32_16x16x32_bf16 (fp32 accumulate), with fp32->bf16
// conversion staged through LDS using packed bf16x2/x4 stores. K-step 64:
// 16 WMMAs per barrier pair per wave. global_prefetch_b8 for next K-tile.
// ---------------------------------------------------------------------------

#define DMODEL 512
#define HEADS  8
#define HDIM   64
#define NLAYERS 6
#define BSZ    4
#define SEQ    512
#define FFD    2048       // WM * D
#define VOUTSZ 32000
#define MTOK   (BSZ * SEQ)   // 2048 rows

typedef __attribute__((ext_vector_type(16))) __bf16 v16bf;
typedef __attribute__((ext_vector_type(8)))  __bf16 v8bf;
typedef __attribute__((ext_vector_type(4)))  __bf16 v4bf;
typedef __attribute__((ext_vector_type(2)))  __bf16 v2bf;
typedef __attribute__((ext_vector_type(8)))  float  v8f;

// A-matrix fragment (16x32 bf16): lanes 0-15 -> K {0..7,16..23},
// lanes 16-31 -> K {8..15,24..31}.  rowbase = start of this lane's M-row.
__device__ __forceinline__ v16bf frag_a(const __bf16* rowbase, int half) {
  const __bf16* p = rowbase + half * 8;
  v8bf lo = *(const v8bf*)(p);
  v8bf hi = *(const v8bf*)(p + 16);
  return __builtin_shufflevector(lo, hi, 0,1,2,3,4,5,6,7,8,9,10,11,12,13,14,15);
}

// B-matrix fragment (32x16 bf16): lanes 0-15 -> K {0..7,16..23},
// lanes 16-31 -> K {8..15,24..31}.  colbase = start of this lane's N-column (k-contiguous storage).
__device__ __forceinline__ v16bf frag_b(const __bf16* colbase, int half) {
  const __bf16* p = colbase + half * 8;
  v8bf lo = *(const v8bf*)(p);
  v8bf hi = *(const v8bf*)(p + 16);
  return __builtin_shufflevector(lo, hi, 0,1,2,3,4,5,6,7,8,9,10,11,12,13,14,15);
}

// ---------------------------------------------------------------------------
// Generic GEMM: C[M,N] = A[M,K] @ W[K,N] + bias[N]   (optional ReLU)
// Block = 256 threads = 8 waves; C-tile 128x128; K-step 64.
// Wave w computes rows [16w,16w+16) x all 128 cols: 2 A-frags reused over
// 8 B-columns -> 16 WMMAs per staging barrier.
// M % 128 == 0, N % 128 == 0, K % 64 == 0 for every call in this model.
// ---------------------------------------------------------------------------
__global__ __launch_bounds__(256) void gemm_kernel(
    const float* __restrict__ A, const float* __restrict__ W,
    const float* __restrict__ bias, float* __restrict__ C,
    int M, int N, int K, int relu)
{
  __shared__ __bf16 As[128][72];   // 72*2B = 144B stride (16B multiple)
  __shared__ __bf16 Wt[128][72];   // W tile stored transposed: Wt[n][k]
  const int tid   = threadIdx.x;
  const int mbase = blockIdx.y * 128;
  const int nbase = blockIdx.x * 128;
  const int wv    = tid >> 5;
  const int lane  = tid & 31;
  const int lm    = lane & 15;
  const int half  = lane >> 4;

  // Staging maps (fixed per thread)
  const int ar  = tid >> 1;          // A row 0..127
  const int ak0 = (tid & 1) * 32;    // A k-group
  const int wn0 = (tid & 15) * 8;    // W n-group (coalesced reads)
  const int wkp = (tid >> 4) * 4;    // W k-group of 4 rows

  const v8f zero = {0.f,0.f,0.f,0.f,0.f,0.f,0.f,0.f};
  v8f acc[8];
#pragma unroll
  for (int t = 0; t < 8; ++t) acc[t] = zero;

  for (int kb = 0; kb < K; kb += 64) {
    // Prefetch next K-tile while this one is staged/computed.
    if (kb + 64 < K) {
      __builtin_prefetch(A + (size_t)(mbase + ar) * K + kb + 64 + ak0, 0, 1);
      __builtin_prefetch(W + (size_t)(kb + 64 + wkp) * N + nbase + wn0, 0, 1);
    }
    // Stage A tile (128x64) fp32 -> bf16, packed v4bf stores (b64).
    {
      const float* ap = A + (size_t)(mbase + ar) * K + kb + ak0;
#pragma unroll
      for (int j = 0; j < 8; ++j) {
        float4 f = ((const float4*)ap)[j];
        v4bf p = {(__bf16)f.x, (__bf16)f.y, (__bf16)f.z, (__bf16)f.w};
        *(v4bf*)&As[ar][ak0 + 4*j] = p;
      }
    }
    // Stage W tile (64x128) transposed -> Wt[n][k]; pack along k (b32 stores).
    {
#pragma unroll
      for (int kk = 0; kk < 4; kk += 2) {
        const float* w0 = W + (size_t)(kb + wkp + kk) * N + nbase + wn0;
        const float* w1 = w0 + N;
        float r0[8], r1[8];
        *(float4*)&r0[0] = ((const float4*)w0)[0];
        *(float4*)&r0[4] = ((const float4*)w0)[1];
        *(float4*)&r1[0] = ((const float4*)w1)[0];
        *(float4*)&r1[4] = ((const float4*)w1)[1];
#pragma unroll
        for (int j = 0; j < 8; ++j) {
          v2bf p = {(__bf16)r0[j], (__bf16)r1[j]};
          *(v2bf*)&Wt[wn0 + j][wkp + kk] = p;
        }
      }
    }
    __syncthreads();

    v16bf a0 = frag_a(&As[16*wv + lm][0],  half);
    v16bf a1 = frag_a(&As[16*wv + lm][32], half);
#pragma unroll
    for (int t = 0; t < 8; ++t) {
      const __bf16* cb = &Wt[16*t + lm][0];
      v16bf b0 = frag_b(cb, half);
      acc[t] = __builtin_amdgcn_wmma_f32_16x16x32_bf16(
          false, a0, false, b0, (short)0, acc[t], false, false);
      v16bf b1 = frag_b(cb + 32, half);
      acc[t] = __builtin_amdgcn_wmma_f32_16x16x32_bf16(
          false, a1, false, b1, (short)0, acc[t], false, false);
    }
    __syncthreads();
  }

#pragma unroll
  for (int t = 0; t < 8; ++t) {
    const int n = nbase + 16*t + lm;
    const float bv = bias[n];
#pragma unroll
    for (int r = 0; r < 8; ++r) {
      const int m = mbase + 16*wv + r + 8*half;
      float v = acc[t][r] + bv;
      if (relu) v = fmaxf(v, 0.f);
      C[(size_t)m * N + n] = v;
    }
  }
}

// ---------------------------------------------------------------------------
// Fused attention for one (batch, head, 64 q-rows): softmax(QK^T/8) @ V.
// 128 threads = 4 waves; wave w owns q-rows [16w,16w+16).
// K/V read as pre-converted bf16 [B,S,512]; head slice is k-contiguous, so K
// rows are directly valid WMMA B-fragments (no transpose for QK^T).
// Dynamic LDS: Q 64x72 bf16 | scores 64x512 f32 (reused in place as bf16 P,
// stride 1024) | Vt 64x40 bf16 | rinv[64].
// ---------------------------------------------------------------------------
#define ATT_SMEM (64*72*2 + 64*512*4 + 64*40*2 + 64*4)

__global__ __launch_bounds__(128) void attn_kernel(
    const float* __restrict__ Q, const __bf16* __restrict__ Kb,
    const __bf16* __restrict__ Vb, float* __restrict__ O)
{
  extern __shared__ __align__(16) char smem[];
  __bf16* q_bf = (__bf16*)smem;                                   // 64 x 72
  float*  sc   = (float*)(smem + 64*72*2);                        // 64 x 512
  __bf16* vt   = (__bf16*)(smem + 64*72*2 + 64*512*4);            // 64 x 40
  float*  rinv = (float*)(smem + 64*72*2 + 64*512*4 + 64*40*2);   // 64

  const int tid  = threadIdx.x;
  const int qt   = blockIdx.x;   // q tile (64 rows)
  const int h    = blockIdx.y;
  const int b    = blockIdx.z;
  const int wv   = tid >> 5;
  const int lane = tid & 31;
  const int lm   = lane & 15;
  const int half = lane >> 4;
  const v8f zero = {0.f,0.f,0.f,0.f,0.f,0.f,0.f,0.f};

  // Load Q tile (64 x 64 head slice) fp32 -> bf16, packed stores.
  {
    const int r  = tid >> 1;
    const int c0 = (tid & 1) * 32;
    const float* qp = Q + ((size_t)b*SEQ + qt*64 + r) * DMODEL + h*HDIM + c0;
#pragma unroll
    for (int j = 0; j < 8; ++j) {
      float4 f = ((const float4*)qp)[j];
      v4bf p = {(__bf16)f.x, (__bf16)f.y, (__bf16)f.z, (__bf16)f.w};
      *(v4bf*)&q_bf[r*72 + c0 + 4*j] = p;
    }
  }
  __syncthreads();

  // Phase 1: scores = Q K^T * 0.125  (K dim = 64 -> 2 WMMA k-chunks)
  v16bf a0 = frag_a(&q_bf[(16*wv + lm) * 72 + 0],  half);
  v16bf a1 = frag_a(&q_bf[(16*wv + lm) * 72 + 32], half);
  for (int nt = 0; nt < SEQ/16; ++nt) {
    const int tok = nt*16 + lm;
    const __bf16* kp = Kb + ((size_t)b*SEQ + tok) * DMODEL + h*HDIM;
    v8f acc = zero;
    v16bf b0 = frag_b(kp, half);
    acc = __builtin_amdgcn_wmma_f32_16x16x32_bf16(false, a0, false, b0, (short)0, acc, false, false);
    v16bf b1 = frag_b(kp + 32, half);
    acc = __builtin_amdgcn_wmma_f32_16x16x32_bf16(false, a1, false, b1, (short)0, acc, false, false);
    const int col = nt*16 + lm;
#pragma unroll
    for (int r = 0; r < 8; ++r)
      sc[(16*wv + r + 8*half) * 512 + col] = acc[r] * 0.125f;
  }
  __syncthreads();

  // Phase 2: row softmax; write bf16 P in place over the fp32 score rows.
  // (ascending c: bf16 slot c lands in fp32 slot c/2 <= c -> no self-clobber)
  __bf16* P = (__bf16*)sc;          // row stride 1024 bf16 elements
  if (tid < 64) {
    const int row = tid;
    float mx = -3.4e38f;
    for (int c = 0; c < SEQ; ++c) {                 // rotated scan (bank spread)
      const int cc = (c + row*8) & (SEQ - 1);
      mx = fmaxf(mx, sc[row*512 + cc]);
    }
    float sum = 0.f;
    for (int c = 0; c < SEQ; ++c) {                 // must be ascending
      const float e = __expf(sc[row*512 + c] - mx);
      sum += e;
      P[row*1024 + c] = (__bf16)e;
    }
    rinv[row] = 1.f / sum;
  }
  __syncthreads();

  // Phase 3: out = P @ V, normalize by 1/sum in epilogue.
  v8f acc2[4] = {zero, zero, zero, zero};
  for (int kc = 0; kc < SEQ/32; ++kc) {
    // Stage V chunk transposed: vt[n][k]; pack k-pairs -> b32 stores.
    {
      const int n0 = (tid & 7) * 8;       // hd column group (coalesced reads)
      const int kp = (tid >> 3) * 2;      // key-position pair 0..30
      const __bf16* vp0 = Vb + ((size_t)b*SEQ + kc*32 + kp) * DMODEL + h*HDIM + n0;
      const __bf16* vp1 = vp0 + DMODEL;
      v8bf r0 = *(const v8bf*)vp0;
      v8bf r1 = *(const v8bf*)vp1;
#pragma unroll
      for (int j = 0; j < 8; ++j) {
        v2bf p = {r0[j], r1[j]};
        *(v2bf*)&vt[(n0 + j)*40 + kp] = p;
      }
    }
    __syncthreads();
    v16bf a = frag_a(&P[(16*wv + lm) * 1024 + kc*32], half);
#pragma unroll
    for (int t = 0; t < 4; ++t) {
      v16bf bb = frag_b(&vt[(16*t + lm) * 40], half);
      acc2[t] = __builtin_amdgcn_wmma_f32_16x16x32_bf16(
          false, a, false, bb, (short)0, acc2[t], false, false);
    }
    __syncthreads();
  }

#pragma unroll
  for (int t = 0; t < 4; ++t) {
#pragma unroll
    for (int r = 0; r < 8; ++r) {
      const int ml = 16*wv + r + 8*half;
      O[((size_t)b*SEQ + qt*64 + ml) * DMODEL + h*HDIM + 16*t + lm] =
          acc2[t][r] * rinv[ml];
    }
  }
}

// ---------------------------------------------------------------------------
// LayerNorm: one wave per row of 512; out = LN(in)*g + b (+ res if addres).
// Population variance (matches jnp.var).
// ---------------------------------------------------------------------------
__global__ __launch_bounds__(32) void ln_kernel(
    float* __restrict__ out, const float* __restrict__ in,
    const float* __restrict__ res, const float* __restrict__ g,
    const float* __restrict__ bb, int addres)
{
  const int row = blockIdx.x, lane = threadIdx.x;
  float v[16];
  float s = 0.f, sq = 0.f;
#pragma unroll
  for (int j = 0; j < 16; ++j) {
    v[j] = in[(size_t)row * DMODEL + j*32 + lane];
    s  += v[j];
    sq += v[j] * v[j];
  }
#pragma unroll
  for (int off = 16; off >= 1; off >>= 1) {
    s  += __shfl_xor(s,  off, 32);
    sq += __shfl_xor(sq, off, 32);
  }
  const float mean = s * (1.f / DMODEL);
  const float var  = sq * (1.f / DMODEL) - mean * mean;
  const float rstd = rsqrtf(var + 1e-5f);
#pragma unroll
  for (int j = 0; j < 16; ++j) {
    const int col = j*32 + lane;
    float o = (v[j] - mean) * rstd * g[col] + bb[col];
    if (addres) o += res[(size_t)row * DMODEL + col];
    out[(size_t)row * DMODEL + col] = o;
  }
}

// ---------------------------------------------------------------------------
// Embedding + sinusoidal positional encoding.
// ---------------------------------------------------------------------------
__global__ __launch_bounds__(128) void embed_kernel(
    float* __restrict__ x, const int* __restrict__ toks,
    const float* __restrict__ emb, int S)
{
  const int rs  = blockIdx.x;          // b*S + s
  const int pos = rs % S;
  const int tok = toks[rs];
#pragma unroll
  for (int j = 0; j < 4; ++j) {
    const int d  = threadIdx.x * 4 + j;
    const int i2 = (d >> 1) * 2;       // 2*(d/2)
    const float f   = __expf(-(float)i2 * (9.210340371976184f / 512.0f)); // 10000^(-2i/D)
    const float ang = (float)pos * f;
    const float pe  = (d & 1) ? __cosf(ang) : __sinf(ang);
    x[(size_t)rs * DMODEL + d] = emb[(size_t)tok * DMODEL + d] + pe;
  }
}

// fp32 -> bf16 tensor conversion (for K/V consumed by attention WMMA)
__global__ __launch_bounds__(256) void cvt_kernel(
    __bf16* __restrict__ o, const float* __restrict__ in, int n)
{
  const int i = (blockIdx.x * 256 + threadIdx.x) * 4;
  if (i < n) {
    float4 f = *(const float4*)(in + i);
    v4bf p = {(__bf16)f.x, (__bf16)f.y, (__bf16)f.z, (__bf16)f.w};
    *(v4bf*)(o + i) = p;
  }
}

// ---------------------------------------------------------------------------
// Host orchestration
// ---------------------------------------------------------------------------
extern "C" void kernel_launch(void* const* d_in, const int* in_sizes, int n_in,
                              void* d_out, int out_size, void* d_ws, size_t ws_size,
                              hipStream_t stream)
{
  (void)in_sizes; (void)n_in; (void)out_size; (void)ws_size;

  const int*   inputs     = (const int*)  d_in[0];
  const int*   outputs    = (const int*)  d_in[1];
  const float* emb_in     = (const float*)d_in[2];
  const float* emb_out    = (const float*)d_in[3];
  const float* enc_attn_w = (const float*)d_in[4];
  const float* enc_attn_b = (const float*)d_in[5];
  const float* enc_ffn_w1 = (const float*)d_in[6];
  const float* enc_ffn_b1 = (const float*)d_in[7];
  const float* enc_ffn_w2 = (const float*)d_in[8];
  const float* enc_ffn_b2 = (const float*)d_in[9];
  const float* enc_ln_g   = (const float*)d_in[10];
  const float* enc_ln_bb  = (const float*)d_in[11];
  const float* dec_self_w = (const float*)d_in[12];
  const float* dec_self_b = (const float*)d_in[13];
  const float* dec_cross_w= (const float*)d_in[14];
  const float* dec_cross_b= (const float*)d_in[15];
  const float* dec_ffn_w1 = (const float*)d_in[16];
  const float* dec_ffn_b1 = (const float*)d_in[17];
  const float* dec_ffn_w2 = (const float*)d_in[18];
  const float* dec_ffn_b2 = (const float*)d_in[19];
  const float* dec_ln_g   = (const float*)d_in[20];
  const float* dec_ln_bb  = (const float*)d_in[21];
  const float* out_w      = (const float*)d_in[22];
  const float* out_b      = (const float*)d_in[23];

  // Workspace layout (~48 MB)
  float* ws = (float*)d_ws;
  const size_t NT = (size_t)MTOK * DMODEL;   // 1M floats per activation tensor
  float* x    = ws;            // encoder stream
  float* y    = ws + 1*NT;     // decoder stream
  float* qbuf = ws + 2*NT;
  float* kbuf = ws + 3*NT;
  float* vbuf = ws + 4*NT;
  float* abuf = ws + 5*NT;     // attention output (pre out-proj)
  float* obuf = ws + 6*NT;     // generic [MTOK, D] temp
  float* hbuf = ws + 7*NT;     // FFN hidden [MTOK, 4D] (4*NT floats)
  __bf16* kbf = (__bf16*)(ws + 11*NT);
  __bf16* vbf = kbf + NT;

  (void)hipFuncSetAttribute(reinterpret_cast<const void*>(attn_kernel),
                            hipFuncAttributeMaxDynamicSharedMemorySize, ATT_SMEM);

  auto gemm = [&](const float* A, const float* W, const float* bias, float* C,
                  int M, int N, int K, int relu) {
    dim3 g(N / 128, M / 128);
    gemm_kernel<<<g, 256, 0, stream>>>(A, W, bias, C, M, N, K, relu);
  };
  auto mha = [&](const float* qin, const float* kvin,
                 const float* W, const float* bv, float* outp) {
    gemm(qin,  W + 0*DMODEL*DMODEL, bv + 0*DMODEL, qbuf, MTOK, DMODEL, DMODEL, 0);
    gemm(kvin, W + 1*DMODEL*DMODEL, bv + 1*DMODEL, kbuf, MTOK, DMODEL, DMODEL, 0);
    gemm(kvin, W + 2*DMODEL*DMODEL, bv + 2*DMODEL, vbuf, MTOK, DMODEL, DMODEL, 0);
    cvt_kernel<<<(int)(NT / 1024), 256, 0, stream>>>(kbf, kbuf, (int)NT);
    cvt_kernel<<<(int)(NT / 1024), 256, 0, stream>>>(vbf, vbuf, (int)NT);
    attn_kernel<<<dim3(SEQ / 64, HEADS, BSZ), 128, ATT_SMEM, stream>>>(qbuf, kbf, vbf, abuf);
    gemm(abuf, W + 3*DMODEL*DMODEL, bv + 3*DMODEL, outp, MTOK, DMODEL, DMODEL, 0);
  };
  auto lnorm = [&](float* outp, const float* inp, const float* resp,
                   const float* g, const float* bb, int addres) {
    ln_kernel<<<MTOK, 32, 0, stream>>>(outp, inp, resp, g, bb, addres);
  };

  embed_kernel<<<MTOK, 128, 0, stream>>>(x, inputs,  emb_in,  SEQ);
  embed_kernel<<<MTOK, 128, 0, stream>>>(y, outputs, emb_out, SEQ);

  // Encoder
  for (int l = 0; l < NLAYERS; ++l) {
    const float* g  = enc_ln_g  + (size_t)l * DMODEL;
    const float* be = enc_ln_bb + (size_t)l * DMODEL;
    mha(x, x, enc_attn_w + (size_t)l*4*DMODEL*DMODEL, enc_attn_b + (size_t)l*4*DMODEL, obuf);
    lnorm(x, obuf, x, g, be, 1);                                   // x = x + LN(attn)
    gemm(x,    enc_ffn_w1 + (size_t)l*DMODEL*FFD, enc_ffn_b1 + (size_t)l*FFD,    hbuf, MTOK, FFD,    DMODEL, 1);
    gemm(hbuf, enc_ffn_w2 + (size_t)l*FFD*DMODEL, enc_ffn_b2 + (size_t)l*DMODEL, obuf, MTOK, DMODEL, FFD,    0);
    lnorm(x, obuf, nullptr, g, be, 0);                             // x = LN(ffn)  (replaces)
  }
  // Decoder
  for (int l = 0; l < NLAYERS; ++l) {
    const float* g  = dec_ln_g  + (size_t)l * DMODEL;
    const float* be = dec_ln_bb + (size_t)l * DMODEL;
    mha(y, y, dec_self_w  + (size_t)l*4*DMODEL*DMODEL, dec_self_b  + (size_t)l*4*DMODEL, obuf);
    lnorm(y, obuf, y, g, be, 1);                                   // y = y + LN(self)
    mha(y, x, dec_cross_w + (size_t)l*4*DMODEL*DMODEL, dec_cross_b + (size_t)l*4*DMODEL, obuf);
    lnorm(y, obuf, y, g, be, 1);                                   // y = y + LN(cross)
    gemm(y,    dec_ffn_w1 + (size_t)l*DMODEL*FFD, dec_ffn_b1 + (size_t)l*FFD,    hbuf, MTOK, FFD,    DMODEL, 1);
    gemm(hbuf, dec_ffn_w2 + (size_t)l*FFD*DMODEL, dec_ffn_b2 + (size_t)l*DMODEL, obuf, MTOK, DMODEL, FFD,    0);
    lnorm(y, obuf, nullptr, g, be, 0);                             // y = LN(ffn)
  }
  // Logits: [2048,512] @ [512,32000] + bias
  gemm(y, out_w, out_b, (float*)d_out, MTOK, VOUTSZ, DMODEL, 0);
}
